// AttnInteractionLayer_16982300688923
// MI455X (gfx1250) — compile-verified, matrix operand order checked by reading, not run on hardware
//
#include <hip/hip_runtime.h>

// R=2048, F=64, D_IN=256, O*H=512. Flattened GEMM: M=131072, K=256, N=512, two
// weight matrices (W_v, W_r). Attention block elided (softmax row-sums == 1):
//   out = LayerNorm(relu(sigmoid(mix)*(x@W_v) + (1-sigmoid(mix))*(x@W_r)))
// bf16 WMMA with f32 accumulation keeps the kernel near the 23.3 TB/s HBM
// roofline (~0.40 GB total traffic); weights stay L2-resident (0.5 MB bf16).

typedef __attribute__((ext_vector_type(16))) __bf16 v16bf;
typedef __attribute__((ext_vector_type(8)))  float  v8f;

#define M_ROWS   131072
#define KDIM     256
#define NDIM     512
#define KCHUNKS  8      // 256 / 32
#define NTILES   32     // 512 / 16
#define LDSP     520    // padded row stride for LDS out buffer

// fp32 -> bf16 round-to-nearest-even (storage only; no __bf16 arithmetic)
__device__ __forceinline__ unsigned short f2bf(float f) {
    unsigned int u = __float_as_uint(f);
    u += 0x7FFFu + ((u >> 16) & 1u);
    return (unsigned short)(u >> 16);
}

// Repack W_v / W_r (fp32 [256,512] row-major) into WMMA B-fragment order, bf16.
// Fragment (mat, ntile, kc): 32 lanes x 16 bf16; lane l covers column
// n = ntile*16 + (l&15); half j holds K = kc*32 + j + 8*((j>>3) + (l>>4)),
// mirroring the ISA 16-bit matrix K-striping. One contiguous 32B per lane.
__global__ void pack_w(const float* __restrict__ Wv,
                       const float* __restrict__ Wr,
                       unsigned short* __restrict__ wpack) {
    int g = blockIdx.x * blockDim.x + threadIdx.x;    // 2*32*8*32 = 16384 units
    if (g >= 2 * NTILES * KCHUNKS * 32) return;
    int mat  = g >> 13;
    int rem  = g & 8191;
    int nt   = rem >> 8;
    int kc   = (rem >> 5) & 7;
    int lane = rem & 31;
    const float* W = mat ? Wr : Wv;
    int n = nt * 16 + (lane & 15);
    unsigned short* dst = wpack + (size_t)g * 16;
#pragma unroll
    for (int j = 0; j < 16; ++j) {
        int k = kc * 32 + j + 8 * ((j >> 3) + (lane >> 4));
        dst[j] = f2bf(W[k * NDIM + n]);
    }
}

__global__ __launch_bounds__(256, 2)
void attn_fused(const float* __restrict__ x,
                const unsigned short* __restrict__ wpack,
                const float* __restrict__ mixp,
                const float* __restrict__ gamma,
                const float* __restrict__ beta,
                float* __restrict__ out) {
    // A stage: 2 row-tiles x 8 kc x 32 lanes x 16 bf16 = 16 KB
    __shared__ __align__(32) unsigned short lds_a[KCHUNKS * 2 * 32 * 16];
    __shared__ float lds_out[16 * LDSP];              // 33 KB, reused per row-tile
    __shared__ float lds_ps[256];
    __shared__ float lds_ps2[256];
    __shared__ float lds_mu[16];
    __shared__ float lds_rs[16];

    const int tid  = threadIdx.x;
    const int lane = tid & 31;
    const int w    = tid >> 5;               // wave id, 0..7 (wave32)
    const long row0 = (long)blockIdx.x * 32; // 32 rows per workgroup

    // ---- Stage A (32 rows x 256 K) into LDS in bf16 fragment layout ----
#pragma unroll
    for (int u = tid; u < 2 * KCHUNKS * 32; u += 256) {
        const int kc = u >> 6;
        const int rt = (u >> 5) & 1;
        const int l  = u & 31;
        const float* xr = x + (row0 + rt * 16 + (l & 15)) * KDIM;
        const int base_k = kc * 32 + 8 * (l >> 4);
        unsigned short* dst = &lds_a[((kc * 2 + rt) * 32 + l) * 16];
#pragma unroll
        for (int j = 0; j < 16; ++j) {
            int k = base_k + j + ((j >= 8) ? 8 : 0);
            dst[j] = f2bf(xr[k]);
        }
    }
    __syncthreads();

    const v16bf* Af = (const v16bf*)lds_a;
    const v16bf* Bp = (const v16bf*)wpack;

    v8f accv0[4] = {}, accv1[4] = {};   // V-GEMM, row-tile 0 / 1
    v8f accr0[4] = {}, accr1[4] = {};   // R-GEMM, row-tile 0 / 1

    // ---- Main loop: 8 K-chunks x 4 N-tiles x {V,R} x 2 row-tiles = 128 WMMA ----
#pragma unroll
    for (int kc = 0; kc < KCHUNKS; ++kc) {
        v16bf a0 = Af[(kc * 2 + 0) * 32 + lane];   // ds_load_b128 x2
        v16bf a1 = Af[(kc * 2 + 1) * 32 + lane];
        if (kc + 1 < KCHUNKS) {
            __builtin_prefetch(&Bp[((w * 4) * KCHUNKS + kc + 1) * 32 + lane], 0, 3);
            __builtin_prefetch(&Bp[((NTILES + w * 4) * KCHUNKS + kc + 1) * 32 + lane], 0, 3);
        }
#pragma unroll
        for (int t = 0; t < 4; ++t) {
            const int nt = w * 4 + t;              // this wave: cols [w*64, w*64+64)
            v16bf bv = Bp[((0 * NTILES + nt) * KCHUNKS + kc) * 32 + lane];
            v16bf br = Bp[((1 * NTILES + nt) * KCHUNKS + kc) * 32 + lane];
            // each B fragment feeds two WMMAs (both row-tiles)
            accv0[t] = __builtin_amdgcn_wmma_f32_16x16x32_bf16(
                false, a0, false, bv, (short)0, accv0[t], false, false);
            accv1[t] = __builtin_amdgcn_wmma_f32_16x16x32_bf16(
                false, a1, false, bv, (short)0, accv1[t], false, false);
            accr0[t] = __builtin_amdgcn_wmma_f32_16x16x32_bf16(
                false, a0, false, br, (short)0, accr0[t], false, false);
            accr1[t] = __builtin_amdgcn_wmma_f32_16x16x32_bf16(
                false, a1, false, br, (short)0, accr1[t], false, false);
        }
    }

    const float mval = 1.0f / (1.0f + expf(-mixp[0]));
    const float nval = 1.0f - mval;

    // ---- Epilogue: two 16-row passes through one LDS buffer ----
#pragma unroll
    for (int rt = 0; rt < 2; ++rt) {
        // mix + relu into LDS (softmax factor == 1, elided)
#pragma unroll
        for (int t = 0; t < 4; ++t) {
            const int col = w * 64 + t * 16 + (lane & 15);
#pragma unroll
            for (int c = 0; c < 8; ++c) {
                const int row = c + 8 * (lane >> 4);   // C-matrix VGPR layout
                const float av = rt ? accv1[t][c] : accv0[t][c];
                const float ar = rt ? accr1[t][c] : accr0[t][c];
                lds_out[row * LDSP + col] = fmaxf(mval * av + nval * ar, 0.0f);
            }
        }
        __syncthreads();

        // LayerNorm over 512 cols per row (biased variance)
        const int row = tid >> 4;
        const int seg = tid & 15;
        const float* rowp = &lds_out[row * LDSP + seg * 32];
        float s = 0.0f, s2 = 0.0f;
#pragma unroll
        for (int i = 0; i < 32; ++i) {
            float v = rowp[i];
            s += v;
            s2 += v * v;
        }
        lds_ps[row * 16 + seg]  = s;
        lds_ps2[row * 16 + seg] = s2;
        __syncthreads();

        if (tid < 16) {
            float ts = 0.0f, ts2 = 0.0f;
#pragma unroll
            for (int i = 0; i < 16; ++i) {
                ts  += lds_ps[tid * 16 + i];
                ts2 += lds_ps2[tid * 16 + i];
            }
            const float mu  = ts * (1.0f / 512.0f);
            const float var = ts2 * (1.0f / 512.0f) - mu * mu;
            lds_mu[tid] = mu;
            lds_rs[tid] = rsqrtf(var + 1e-5f);
        }
        __syncthreads();

        const float mu = lds_mu[row];
        const float rs = lds_rs[row];
        float* outp = out + (size_t)(row0 + rt * 16 + row) * NDIM + seg * 32;
        const float* gp = gamma + seg * 32;
        const float* bp = beta + seg * 32;
#pragma unroll
        for (int i = 0; i < 32; i += 4) {
            float4 y;
            y.x = (rowp[i + 0] - mu) * rs * gp[i + 0] + bp[i + 0];
            y.y = (rowp[i + 1] - mu) * rs * gp[i + 1] + bp[i + 1];
            y.z = (rowp[i + 2] - mu) * rs * gp[i + 2] + bp[i + 2];
            y.w = (rowp[i + 3] - mu) * rs * gp[i + 3] + bp[i + 3];
            *(float4*)(outp + i) = y;
        }
        __syncthreads();   // buffer reused by next row-tile
    }
}

extern "C" void kernel_launch(void* const* d_in, const int* in_sizes, int n_in,
                              void* d_out, int out_size, void* d_ws, size_t ws_size,
                              hipStream_t stream) {
    // setup_inputs order: x, W_q, W_k, W_v, W_r, mix, gamma, beta
    const float* x     = (const float*)d_in[0];
    const float* Wv    = (const float*)d_in[3];
    const float* Wr    = (const float*)d_in[4];
    const float* mix   = (const float*)d_in[5];
    const float* gamma = (const float*)d_in[6];
    const float* beta  = (const float*)d_in[7];
    float* out = (float*)d_out;

    unsigned short* wpack = (unsigned short*)d_ws; // 2*256*512*2 B = 512 KB

    pack_w<<<64, 256, 0, stream>>>(Wv, Wr, wpack);
    attn_fused<<<M_ROWS / 32, 256, 0, stream>>>(x, wpack, mix, gamma, beta, out);
}
